// SimpleCharRNN_4887672783567
// MI455X (gfx1250) — compile-verified, matrix operand order checked by reading, not run on hardware
//
#include <hip/hip_runtime.h>
#include <hip/hip_bf16.h>

// ---------------------------------------------------------------------------
// Persistent batch-sliced LSTM for MI455X (gfx1250, wave32, WMMA bf16).
//   grid = 32 workgroups (B/16), block = 512 threads (16 waves, 4/SIMD).
//   Each workgroup owns 16 batch rows for all T=1024 steps (no grid sync).
//   Wave w owns 16 hidden cols -> 4 gate N-tiles; per step 40 WMMAs/wave
//   (v_wmma_f32_16x16x32_bf16), pointwise LSTM in accumulator layout,
//   h -> LDS, h2o projection via 8 more WMMAs on wave 0.
//   Packed weights live in L2 (640 KB); an opaque per-step SGPR offset keeps
//   the fragment loads inside the time loop (prevents LICM -> scratch spill).
// ---------------------------------------------------------------------------

typedef __attribute__((ext_vector_type(16))) __bf16 v16bf;
typedef __attribute__((ext_vector_type(8)))  __bf16 v8bf;
typedef __attribute__((ext_vector_type(8)))  float  v8f;

#define TT   1024
#define DD   55
#define HH   256
#define OO   9
#define KTOT 320          // 55 + 9 + 256
#define NTL  64           // 1024/16 gate n-tiles
#define KT   10           // 320/32 k-tiles
#define STR  328          // LDS row stride (halfs), 16B aligned, bank-skewed
#define NTHR 512          // threads per block (16 waves)
#define WCAT_ELEMS (NTL * KT * 32 * 16)  // 327680 bf16 = 640 KB
#define WHO_ELEMS  (8 * 32 * 16)         // 4096 bf16   = 8 KB

// Fast device transcendentals: v_exp_f32 / v_rcp_f32 / v_tanh_f32.
static __device__ __forceinline__ float sigmf(float x) {
    return __builtin_amdgcn_rcpf(1.0f + __expf(-x));
}
static __device__ __forceinline__ float tanhf_fast(float x) {
#if __has_builtin(__builtin_amdgcn_tanhf)
    return __builtin_amdgcn_tanhf(x);
#else
    // 1 - 2/(1+e^{2x}) : overflow-safe at both ends
    return 1.0f - 2.0f * __builtin_amdgcn_rcpf(1.0f + __expf(2.0f * x));
#endif
}

// Build a 16x32 bf16 A-fragment for one lane from an LDS row (row-major).
// Lanes 0-15 hold K 0..7,16..23; lanes 16-31 hold K 8..15,24..31.
static __device__ __forceinline__ v16bf load_a_frag(const __bf16* row, int kbase, int lane) {
    int o = (lane & 16) ? 8 : 0;
    v8bf lo = *(const v8bf*)(row + kbase + o);
    v8bf hi = *(const v8bf*)(row + kbase + 16 + o);
    return __builtin_shufflevector(lo, hi, 0,1,2,3,4,5,6,7,8,9,10,11,12,13,14,15);
}

// ---------------------------------------------------------------------------
// Prepack W_cat = [W_ih | W_hh] (4H x 320) and zero-padded W_ho^T (256 x 16)
// into per-lane bf16 WMMA B-fragment layout in workspace.
// ---------------------------------------------------------------------------
__global__ __launch_bounds__(256) void lstm_prepack(
    const float* __restrict__ W_ih,   // [1024 x 64]
    const float* __restrict__ W_hh,   // [1024 x 256]
    const float* __restrict__ W_ho,   // [9 x 256]
    __bf16* __restrict__ ws)
{
    int id = blockIdx.x * 256 + threadIdx.x;
    if (id < WCAT_ELEMS) {
        int e     = id & 15;
        int lane  = (id >> 4) & 31;
        int kt    = (id >> 9) % KT;
        int ntile = id / (KT * 512);
        int n  = ntile * 16 + (lane & 15);
        int kl = (e < 8 ? e : e + 8) + ((lane & 16) ? 8 : 0);
        int k  = kt * 32 + kl;
        float v = (k < 64) ? W_ih[n * 64 + k] : W_hh[n * 256 + (k - 64)];
        ws[id] = (__bf16)v;
    } else if (id < WCAT_ELEMS + WHO_ELEMS) {
        int id2  = id - WCAT_ELEMS;
        int e    = id2 & 15;
        int lane = (id2 >> 4) & 31;
        int kt   = id2 >> 9;              // 0..7
        int j    = lane & 15;             // output column (pad 9..15 with 0)
        int kl   = (e < 8 ? e : e + 8) + ((lane & 16) ? 8 : 0);
        int k    = kt * 32 + kl;
        float v  = (j < OO) ? W_ho[j * 256 + k] : 0.0f;
        ws[WCAT_ELEMS + id2] = (__bf16)v;
    }
}

// ---------------------------------------------------------------------------
// Main persistent kernel.
// ---------------------------------------------------------------------------
__global__ __launch_bounds__(NTHR) void lstm_persist(
    const float* __restrict__ x,      // [B,T,55]
    const float* __restrict__ h0,     // [B,256]
    const float* __restrict__ c0,     // [B,256]
    const float* __restrict__ b_ih,   // [1024]
    const float* __restrict__ b_hh,   // [1024]
    const float* __restrict__ b_ho,   // [9]
    const __bf16* __restrict__ wsb,   // packed weights
    float* __restrict__ out)          // [B,T,9]
{
    __shared__ __align__(16) __bf16 sbuf[2][16 * STR];  // [x(55)|out(9)|h(256)]
    __shared__ float sbias[1024];
    __shared__ float sbho[16];

    const int tid  = threadIdx.x;
    const int lane = tid & 31;
    const int w    = tid >> 5;          // wave id 0..15 -> h cols [16w,16w+16)
    const int b0   = blockIdx.x * 16;   // batch slice base

    // ---- one-time staging ----
    for (int i = tid; i < 1024; i += NTHR) sbias[i] = b_ih[i] + b_hh[i];
    if (tid < 16) sbho[tid] = (tid < OO) ? b_ho[tid] : 0.0f;

    for (int i = tid; i < 16 * DD; i += NTHR) {
        int m = i / DD, d = i - m * DD;
        sbuf[0][m * STR + d] = (__bf16)x[((size_t)(b0 + m) * TT) * DD + d];
    }
    for (int i = tid; i < 16 * OO; i += NTHR) {
        int m = i / OO, j = i - m * OO;
        sbuf[0][m * STR + DD + j] = (__bf16)0.0f;
    }
    for (int i = tid; i < 16 * HH; i += NTHR) {
        int m = i >> 8, k = i & 255;
        sbuf[0][m * STR + 64 + k] = (__bf16)h0[(size_t)(b0 + m) * HH + k];
    }

    // cell state in registers, WMMA accumulator layout (one n-tile per wave)
    const int colh = w * 16 + (lane & 15);
    float cp[8];
#pragma unroll
    for (int r = 0; r < 8; ++r) {
        int m = r + ((lane & 16) ? 8 : 0);
        cp[r] = c0[(size_t)(b0 + m) * HH + colh];
    }
    __syncthreads();

    const v16bf* WB  = (const v16bf*)wsb;
    const v16bf* WB2 = (const v16bf*)(wsb + WCAT_ELEMS);

    for (int t = 0; t < TT; ++t) {
        // Opaque zero: same address every step, but not provably loop-invariant.
        // Keeps B-fragment loads inside the t-loop (streamed from L2) instead
        // of LICM hoist -> spill -> scratch reloads.
        int tofs;
        asm volatile("s_mov_b32 %0, 0" : "=s"(tofs));
        const v16bf* WBt  = WB  + tofs;
        const v16bf* WB2t = WB2 + tofs;

        const int cur = t & 1, nxt = cur ^ 1;
        const __bf16* arow = &sbuf[cur][(lane & 15) * STR];

        v8f acc[4];
#pragma unroll
        for (int g = 0; g < 4; ++g) acc[g] = (v8f)0.0f;

        // gates GEMM: [16,320] x [320,64(this wave)] -> 4 accumulators
#pragma unroll 1
        for (int kt = 0; kt < KT; ++kt) {
            v16bf a = load_a_frag(arow, kt * 32, lane);
            v16bf bfr[4];
#pragma unroll
            for (int g = 0; g < 4; ++g) {
                int ntile = g * 16 + w;
                bfr[g] = WBt[(ntile * KT + kt) * 32 + lane];
            }
#pragma unroll
            for (int g = 0; g < 4; ++g) {
                acc[g] = __builtin_amdgcn_wmma_f32_16x16x32_bf16(
                    false, a, false, bfr[g], (short)0, acc[g], false, false);
            }
        }

        // pointwise LSTM in accumulator layout; h -> LDS(nxt) as bf16
        {
            float bi  = sbias[colh];
            float bff = sbias[256 + colh];
            float bg  = sbias[512 + colh];
            float bo  = sbias[768 + colh];
#pragma unroll
            for (int r = 0; r < 8; ++r) {
                float iv = sigmf(acc[0][r] + bi);
                float fv = sigmf(acc[1][r] + bff);
                float gv = tanhf_fast(acc[2][r] + bg);
                float ov = sigmf(acc[3][r] + bo);
                float c  = fv * cp[r] + iv * gv;
                cp[r] = c;
                float h  = ov * tanhf_fast(c);
                int m = r + ((lane & 16) ? 8 : 0);
                sbuf[nxt][m * STR + 64 + colh] = (__bf16)h;
            }
        }
        __syncthreads();   // all h_t written before h2o / next-step reads

        if (w == 0) {
            // out = h @ W_ho^T + b_ho via one WMMA n-tile (cols 9..15 padded)
            const __bf16* hrow = &sbuf[nxt][(lane & 15) * STR];
            v8f a2 = (v8f)0.0f;
#pragma unroll
            for (int kt = 0; kt < 8; ++kt) {
                v16bf a  = load_a_frag(hrow, 64 + kt * 32, lane);
                v16bf bf = WB2t[kt * 32 + lane];
                a2 = __builtin_amdgcn_wmma_f32_16x16x32_bf16(
                    false, a, false, bf, (short)0, a2, false, false);
            }
            int j = lane & 15;
            if (j < OO) {
                float bb = sbho[j];
#pragma unroll
                for (int r = 0; r < 8; ++r) {
                    int m = r + ((lane & 16) ? 8 : 0);
                    float v = a2[r] + bb;
                    out[((size_t)(b0 + m) * TT + t) * OO + j] = v;
                    sbuf[nxt][m * STR + DD + j] = (__bf16)v;   // out_prev for t+1
                }
            }
        } else if (t + 1 < TT) {
            // waves 1..15 stage x_{t+1} into nxt buffer (read-once -> NT loads)
            for (int i = tid - 32; i < 16 * DD; i += (NTHR - 32)) {
                int m = i / DD, d = i - m * DD;
                float xv = __builtin_nontemporal_load(
                    &x[((size_t)(b0 + m) * TT + (t + 1)) * DD + d]);
                sbuf[nxt][m * STR + d] = (__bf16)xv;
            }
        }
        __syncthreads();   // nxt buffer complete for next iteration
    }
}

// ---------------------------------------------------------------------------
extern "C" void kernel_launch(void* const* d_in, const int* in_sizes, int n_in,
                              void* d_out, int out_size, void* d_ws, size_t ws_size,
                              hipStream_t stream) {
    (void)in_sizes; (void)n_in; (void)out_size; (void)ws_size;
    const float* x    = (const float*)d_in[0];
    // d_in[1] = lengths (unused by the reference's scan)
    const float* h0   = (const float*)d_in[2];
    const float* c0   = (const float*)d_in[3];
    const float* W_ih = (const float*)d_in[4];
    const float* b_ih = (const float*)d_in[5];
    const float* W_hh = (const float*)d_in[6];
    const float* b_hh = (const float*)d_in[7];
    const float* W_ho = (const float*)d_in[8];
    const float* b_ho = (const float*)d_in[9];
    __bf16* wsb = (__bf16*)d_ws;   // needs ~664 KB

    int prep_threads = WCAT_ELEMS + WHO_ELEMS;          // 331776
    lstm_prepack<<<(prep_threads + 255) / 256, 256, 0, stream>>>(W_ih, W_hh, W_ho, wsb);
    lstm_persist<<<32, NTHR, 0, stream>>>(x, h0, c0, b_ih, b_hh, b_ho, wsb,
                                          (float*)d_out);
}
